// ToMeAttention_35974646071452
// MI455X (gfx1250) — matrix operand order repaired and checked by reading, not compiled, not run on hardware
//
#include <hip/hip_runtime.h>
#include <math.h>

typedef __attribute__((ext_vector_type(16))) __bf16 v16bf;
typedef __attribute__((ext_vector_type(8)))  float  v8f;

#define B_  8
#define T_  512
#define D_  1024
#define H_  16
#define HD_ 64
#define TA_ 256   // even tokens
#define TB_ 256   // odd tokens
#define TM_ 384   // merged length = TA + (TB - R)
#define R_  128

// ---------------- WMMA helpers (CDNA5 layouts, ISA 7.12.2) ----------------

__device__ __forceinline__ v8f wmma_bf16(v16bf a, v16bf b, v8f c) {
  return __builtin_amdgcn_wmma_f32_16x16x32_bf16(
      false, a, false, b, (short)0, c, false, false);
}

// A fragment: 16x32 (MxK) bf16, row-major source with leading dim ld.
// lane<16  : M=lane,    e=0..15 -> K=e
// lane>=16 : M=lane-16, e=0..15 -> K=e+16
__device__ __forceinline__ v16bf load_A_frag(const __bf16* A, int ld, int lane) {
  int half = lane >> 4;
  int m    = lane & 15;
  const __bf16* row = A + (size_t)m * ld;
  int k0 = 16 * half;
  int k1 = 16 * half + 8;
  v16bf a;
#pragma unroll
  for (int e = 0; e < 8; ++e) a[e] = row[k0 + e];
#pragma unroll
  for (int e = 0; e < 8; ++e) a[8 + e] = row[k1 + e];
  return a;
}

// B fragment: 32x16 (KxN). Source stored as N rows x K cols (row-major, ld).
// lane -> N = lane&15, K base = 16*(lane>>4); e -> K = base+e (contiguous).
__device__ __forceinline__ v16bf load_B_frag_nk(const __bf16* Bm, int ld, int lane) {
  int n  = lane & 15;
  int kb = 16 * (lane >> 4);
  const __bf16* row = Bm + (size_t)n * ld + kb;
  v16bf b;
#pragma unroll
  for (int e = 0; e < 16; ++e) b[e] = row[e];
  return b;
}

__device__ __forceinline__ float rowmax16(float v) {
  v = fmaxf(v, __shfl_xor(v, 1));
  v = fmaxf(v, __shfl_xor(v, 2));
  v = fmaxf(v, __shfl_xor(v, 4));
  v = fmaxf(v, __shfl_xor(v, 8));
  return v;
}
__device__ __forceinline__ float rowsum16(float v) {
  v += __shfl_xor(v, 1);
  v += __shfl_xor(v, 2);
  v += __shfl_xor(v, 4);
  v += __shfl_xor(v, 8);
  return v;
}

// ---------------- 1. f32 -> bf16 convert (weights) ----------------

__global__ void k_f32_to_bf16(const float* __restrict__ src, __bf16* __restrict__ dst, int n) {
  int i = (blockIdx.x * blockDim.x + threadIdx.x) * 4;
#pragma unroll
  for (int j = 0; j < 4; ++j)
    if (i + j < n) dst[i + j] = (__bf16)src[i + j];
}

// ---------------- 2. token normalize -> bf16 (an, bn) ----------------

__global__ void k_norm(const float* __restrict__ x, __bf16* __restrict__ an,
                       __bf16* __restrict__ bn) {
  int rid = blockIdx.x;            // b*512 + tk
  int b = rid >> 9, tk = rid & 511;
  int tid = threadIdx.x;
  const float* src = x + ((size_t)(b * T_ + tk)) * D_;
  __shared__ float red[256];
  float s = 0.f;
  for (int i = tid; i < D_; i += 256) { float v = src[i]; s += v * v; }
  red[tid] = s;
  __syncthreads();
  for (int st = 128; st > 0; st >>= 1) {
    if (tid < st) red[tid] += red[tid + st];
    __syncthreads();
  }
  float nrm = sqrtf(red[0]);
  float sc = 1.0f / fmaxf(nrm, 1e-12f);
  __bf16* dst = (tk & 1) ? (bn + ((size_t)(b * TB_ + (tk >> 1))) * D_)
                         : (an + ((size_t)(b * TA_ + (tk >> 1))) * D_);
  for (int i = tid; i < D_; i += 256) dst[i] = (__bf16)(src[i] * sc);
}

// ---------------- 3. cosine-sim GEMM: sim[b] = an @ bn^T (WMMA) ----------------

__global__ void k_sim(const __bf16* __restrict__ an, const __bf16* __restrict__ bn,
                      float* __restrict__ sim) {
  int b = blockIdx.z, tm = blockIdx.x, tn = blockIdx.y;
  int lane = threadIdx.x & 31;
  const __bf16* A  = an + ((size_t)b * TA_ + tm * 16) * D_;
  const __bf16* Bm = bn + ((size_t)b * TB_ + tn * 16) * D_;
  v8f acc = {};
  for (int kc = 0; kc < D_; kc += 32) {
    v16bf a  = load_A_frag(A + kc, D_, lane);
    v16bf bb = load_B_frag_nk(Bm + kc, D_, lane);
    acc = wmma_bf16(a, bb, acc);
  }
  int half = lane >> 4, N = lane & 15;
  float* out = sim + (size_t)b * (TA_ * TB_);
#pragma unroll
  for (int r = 0; r < 8; ++r) {
    int M = r + 8 * half;
    out[(tm * 16 + M) * TB_ + tn * 16 + N] = acc[r];
  }
}

// ---------------- 4. greedy matching (== sorted-greedy) ----------------

__global__ void k_match(const float* __restrict__ sim, int* __restrict__ pa,
                        int* __restrict__ pb, int* __restrict__ kept,
                        int* __restrict__ bmap) {
  int b = blockIdx.x, t = threadIdx.x;
  __shared__ int   used_a[256], used_b[256];
  __shared__ float rv[256];
  __shared__ int   ri[256];
  __shared__ int   pa_l[128], pb_l[128];
  used_a[t] = 0; used_b[t] = 0;
  __syncthreads();
  const float* S = sim + (size_t)b * (TA_ * TB_);
  for (int it = 0; it < R_; ++it) {
    float mv = -3.4e38f; int mi = -1;
    if (!used_a[t]) {
      const float* row = S + t * TB_;
      for (int c = 0; c < TB_; ++c) {
        if (!used_b[c]) {
          float v = row[c];
          if (v > mv) { mv = v; mi = t * TB_ + c; }   // '>' keeps smallest idx
        }
      }
    }
    rv[t] = mv; ri[t] = mi;
    __syncthreads();
    for (int st = 128; st > 0; st >>= 1) {
      if (t < st) {
        float v2 = rv[t + st]; int i2 = ri[t + st];
        bool better = (v2 > rv[t]) ||
                      (v2 == rv[t] && i2 >= 0 && (ri[t] < 0 || i2 < ri[t]));
        if (better) { rv[t] = v2; ri[t] = i2; }
      }
      __syncthreads();
    }
    if (t == 0) {
      int fi = ri[0];
      int ai = fi >> 8, ci = fi & 255;
      pa_l[it] = ai; pb_l[it] = ci;
      used_a[ai] = 1; used_b[ci] = 1;
      pa[b * R_ + it] = ai; pb[b * R_ + it] = ci;
    }
    __syncthreads();
  }
  if (t == 0) {
    int cnt = 0;
    for (int c = 0; c < TB_; ++c)
      if (!used_b[c]) { kept[b * R_ + cnt] = c; bmap[b * TB_ + c] = TA_ + cnt; cnt++; }
    for (int i = 0; i < R_; ++i) bmap[b * TB_ + pb_l[i]] = pa_l[i];
  }
}

// ---------------- 5. build merged sequence (bf16) ----------------

__global__ void k_build_merged(const float* __restrict__ x, const int* __restrict__ kept,
                               __bf16* __restrict__ merged) {
  int b = blockIdx.y, t = blockIdx.x, tid = threadIdx.x;
  const float* src;
  if (t < TA_) src = x + ((size_t)(b * T_ + 2 * t)) * D_;
  else {
    int c = kept[b * R_ + (t - TA_)];
    src = x + ((size_t)(b * T_ + 2 * c + 1)) * D_;
  }
  __bf16* dst = merged + ((size_t)(b * TM_ + t)) * D_;
  for (int c = tid; c < D_; c += 256) dst[c] = (__bf16)src[c];
}

__global__ void k_merge_pairs(const float* __restrict__ x, const int* __restrict__ pa,
                              const int* __restrict__ pb, __bf16* __restrict__ merged) {
  int b = blockIdx.y, i = blockIdx.x, tid = threadIdx.x;
  int ai = pa[b * R_ + i], bi = pb[b * R_ + i];
  const float* xa = x + ((size_t)(b * T_ + 2 * ai)) * D_;
  const float* xb = x + ((size_t)(b * T_ + 2 * bi + 1)) * D_;
  __bf16* dst = merged + ((size_t)(b * TM_ + ai)) * D_;
  for (int c = tid; c < D_; c += 256) dst[c] = (__bf16)(0.5f * (xa[c] + xb[c]));
}

// ---------------- 6. GEMM: Y = X @ W^T (WMMA, wave computes 64x32 tile) ----------------
// 4 M-subtiles x 2 N-subtiles per wave: 8 WMMA per k-step, A reused 2x, B 4x.
// mode 0: f32 row-major Y[rows][D]                         (dstF)
// mode 1: bf16 head layout [b][h][t][hd]                   (dstB)  Q/K
// mode 2: bf16 transposed head layout [b][h][hd][t]        (dstB)  V

#define GM_ 4
#define GN_ 2

__global__ void k_gemm(const __bf16* __restrict__ X, const __bf16* __restrict__ W,
                       float* __restrict__ dstF, __bf16* __restrict__ dstB,
                       int mode, float scale) {
  int tm = blockIdx.x, tn = blockIdx.y;
  int lane = threadIdx.x & 31;
  const __bf16* Abase = X + ((size_t)tm * (16 * GM_)) * D_;
  const __bf16* Bbase = W + ((size_t)tn * (16 * GN_)) * D_;
  v8f acc[GM_][GN_];
#pragma unroll
  for (int i = 0; i < GM_; ++i)
#pragma unroll
    for (int j = 0; j < GN_; ++j) acc[i][j] = (v8f){};

  for (int kc = 0; kc < D_; kc += 32) {
    v16bf a[GM_], bb[GN_];
#pragma unroll
    for (int i = 0; i < GM_; ++i)
      a[i] = load_A_frag(Abase + (size_t)i * 16 * D_ + kc, D_, lane);
#pragma unroll
    for (int j = 0; j < GN_; ++j)
      bb[j] = load_B_frag_nk(Bbase + (size_t)j * 16 * D_ + kc, D_, lane);
#pragma unroll
    for (int i = 0; i < GM_; ++i)
#pragma unroll
      for (int j = 0; j < GN_; ++j)
        acc[i][j] = wmma_bf16(a[i], bb[j], acc[i][j]);
  }

  int half = lane >> 4, N = lane & 15;
#pragma unroll
  for (int i = 0; i < GM_; ++i) {
#pragma unroll
    for (int j = 0; j < GN_; ++j) {
#pragma unroll
      for (int r = 0; r < 8; ++r) {
        int row = tm * (16 * GM_) + i * 16 + r + 8 * half;
        int col = tn * (16 * GN_) + j * 16 + N;
        float val = acc[i][j][r] * scale;
        if (mode == 0) {
          dstF[(size_t)row * D_ + col] = val;
        } else {
          int b = row / TM_, t = row % TM_;
          int h = col >> 6, d = col & 63;
          size_t idx = (mode == 1)
              ? (((size_t)b * H_ + h) * TM_ + t) * HD_ + d
              : (((size_t)b * H_ + h) * HD_ + d) * TM_ + t;
          dstB[idx] = (__bf16)val;
        }
      }
    }
  }
}

// ---------------- 7. flash attention (WMMA, online softmax) ----------------
// Grid (TM/128, H, B); block 256 = 8 waves; each wave owns 16 q-rows.
// V is pre-transposed [b][h][d][t] so P@V uses the contiguous nk B path.

__global__ void k_attn(const __bf16* __restrict__ Q, const __bf16* __restrict__ Km,
                       const __bf16* __restrict__ VT, __bf16* __restrict__ O) {
  int qblk = blockIdx.x, h = blockIdx.y, b = blockIdx.z;
  int wave = threadIdx.x >> 5, lane = threadIdx.x & 31;
  int half = lane >> 4, N = lane & 15;
  int qbase = qblk * 128 + wave * 16;
  size_t base = ((size_t)b * H_ + h) * TM_ * HD_;   // same total size for Q/K and VT

  __shared__ __bf16 plds[8][16 * 32];

  // Q fragments for K-dim 0..31 and 32..63 (loaded once)
  v16bf qa0 = load_A_frag(Q + base + (size_t)qbase * HD_ + 0,  HD_, lane);
  v16bf qa1 = load_A_frag(Q + base + (size_t)qbase * HD_ + 32, HD_, lane);

  float m_[8], l_[8];
  v8f acc0 = {}, acc1 = {}, acc2 = {}, acc3 = {};
#pragma unroll
  for (int r = 0; r < 8; ++r) { m_[r] = -1e30f; l_[r] = 0.f; }

  for (int kb0 = 0; kb0 < TM_; kb0 += 32) {
    // scores for 32 keys: two 16-wide n-tiles, each K=64 via 2 WMMA
    v8f s0 = {}, s1 = {};
    {
      const __bf16* K0 = Km + base + (size_t)kb0 * HD_;
      v16bf b0 = load_B_frag_nk(K0 + 0,  HD_, lane);
      v16bf b1 = load_B_frag_nk(K0 + 32, HD_, lane);
      s0 = wmma_bf16(qa0, b0, s0);
      s0 = wmma_bf16(qa1, b1, s0);
      const __bf16* K1 = Km + base + (size_t)(kb0 + 16) * HD_;
      v16bf b2 = load_B_frag_nk(K1 + 0,  HD_, lane);
      v16bf b3 = load_B_frag_nk(K1 + 32, HD_, lane);
      s1 = wmma_bf16(qa0, b2, s1);
      s1 = wmma_bf16(qa1, b3, s1);
    }
    float p0[8], p1[8];
#pragma unroll
    for (int r = 0; r < 8; ++r) {
      float smax = rowmax16(fmaxf(s0[r], s1[r]));
      float mnew = fmaxf(m_[r], smax);
      float alpha = __expf(m_[r] - mnew);
      float e0 = __expf(s0[r] - mnew);
      float e1 = __expf(s1[r] - mnew);
      float rs = rowsum16(e0 + e1);
      l_[r] = l_[r] * alpha + rs;
      m_[r] = mnew;
      acc0[r] *= alpha; acc1[r] *= alpha; acc2[r] *= alpha; acc3[r] *= alpha;
      p0[r] = e0; p1[r] = e1;
    }
    // repack probabilities (C layout) into A-fragment layout via LDS
    __bf16* pw = &plds[wave][0];
#pragma unroll
    for (int r = 0; r < 8; ++r) {
      int M = r + 8 * half;
      pw[M * 32 + N]      = (__bf16)p0[r];
      pw[M * 32 + 16 + N] = (__bf16)p1[r];
    }
    __syncthreads();
    v16bf pA = load_A_frag(pw, 32, lane);
    // P (16x32) @ V (32x64): 4 n-tiles; VT rows are head-dims, keys inner
    const __bf16* Vb = VT + base + kb0;   // element (d, t=kb0)
    acc0 = wmma_bf16(pA, load_B_frag_nk(Vb + (size_t) 0 * TM_, TM_, lane), acc0);
    acc1 = wmma_bf16(pA, load_B_frag_nk(Vb + (size_t)16 * TM_, TM_, lane), acc1);
    acc2 = wmma_bf16(pA, load_B_frag_nk(Vb + (size_t)32 * TM_, TM_, lane), acc2);
    acc3 = wmma_bf16(pA, load_B_frag_nk(Vb + (size_t)48 * TM_, TM_, lane), acc3);
    __syncthreads();
  }
  // write O[b][t][h*64 + d] (bf16, model-dim inner => ready for Wo GEMM)
#pragma unroll
  for (int r = 0; r < 8; ++r) {
    int t = qbase + r + 8 * half;
    float inv = 1.0f / l_[r];
    size_t o = ((size_t)b * TM_ + t) * D_ + h * HD_;
    O[o +  0 + N] = (__bf16)(acc0[r] * inv);
    O[o + 16 + N] = (__bf16)(acc1[r] * inv);
    O[o + 32 + N] = (__bf16)(acc2[r] * inv);
    O[o + 48 + N] = (__bf16)(acc3[r] * inv);
  }
}

// ---------------- 8. unmerge scatter ----------------

__global__ void k_unmerge(const float* __restrict__ Y, const int* __restrict__ bmap,
                          float* __restrict__ out) {
  int b = blockIdx.y, tk = blockIdx.x, tid = threadIdx.x;
  int src = ((tk & 1) == 0) ? (tk >> 1) : bmap[b * TB_ + (tk >> 1)];
  const float* s = Y + ((size_t)(b * TM_ + src)) * D_;
  float* d = out + ((size_t)(b * T_ + tk)) * D_;
  for (int c = tid; c < D_; c += 256) d[c] = s[c];
}

// ---------------- launcher ----------------

extern "C" void kernel_launch(void* const* d_in, const int* in_sizes, int n_in,
                              void* d_out, int out_size, void* d_ws, size_t ws_size,
                              hipStream_t stream) {
  const float* x  = (const float*)d_in[0];
  const float* Wq = (const float*)d_in[1];
  const float* Wk = (const float*)d_in[2];
  const float* Wv = (const float*)d_in[3];
  const float* Wo = (const float*)d_in[4];
  float* out = (float*)d_out;

  char* ws = (char*)d_ws;
  size_t off = 0;
  auto alloc = [&](size_t bytes) { size_t o = off; off += (bytes + 255) & ~(size_t)255; return o; };

  __bf16* an_bf  = (__bf16*)(ws + alloc((size_t)B_ * TA_ * D_ * 2));
  __bf16* bn_bf  = (__bf16*)(ws + alloc((size_t)B_ * TB_ * D_ * 2));
  __bf16* Wq_bf  = (__bf16*)(ws + alloc((size_t)D_ * D_ * 2));
  __bf16* Wk_bf  = (__bf16*)(ws + alloc((size_t)D_ * D_ * 2));
  __bf16* Wv_bf  = (__bf16*)(ws + alloc((size_t)D_ * D_ * 2));
  __bf16* Wo_bf  = (__bf16*)(ws + alloc((size_t)D_ * D_ * 2));
  float*  sim    = (float*) (ws + alloc((size_t)B_ * TA_ * TB_ * 4));
  int*    pa     = (int*)   (ws + alloc((size_t)B_ * R_ * 4));
  int*    pb     = (int*)   (ws + alloc((size_t)B_ * R_ * 4));
  int*    kept   = (int*)   (ws + alloc((size_t)B_ * R_ * 4));
  int*    bmap   = (int*)   (ws + alloc((size_t)B_ * TB_ * 4));
  __bf16* merged = (__bf16*)(ws + alloc((size_t)B_ * TM_ * D_ * 2));
  __bf16* Qbf    = (__bf16*)(ws + alloc((size_t)B_ * H_ * TM_ * HD_ * 2));
  __bf16* Kbf    = (__bf16*)(ws + alloc((size_t)B_ * H_ * TM_ * HD_ * 2));
  __bf16* VTbf   = (__bf16*)(ws + alloc((size_t)B_ * H_ * HD_ * TM_ * 2));
  __bf16* Obf    = (__bf16*)(ws + alloc((size_t)B_ * TM_ * D_ * 2));
  float*  Yf     = (float*) (ws + alloc((size_t)B_ * TM_ * D_ * 4));
  (void)ws_size; (void)in_sizes; (void)n_in; (void)out_size;

  // 1. weights -> bf16
  {
    int n = D_ * D_;
    dim3 g((n + 1023) / 1024), blk(256);
    k_f32_to_bf16<<<g, blk, 0, stream>>>(Wq, Wq_bf, n);
    k_f32_to_bf16<<<g, blk, 0, stream>>>(Wk, Wk_bf, n);
    k_f32_to_bf16<<<g, blk, 0, stream>>>(Wv, Wv_bf, n);
    k_f32_to_bf16<<<g, blk, 0, stream>>>(Wo, Wo_bf, n);
  }
  // 2. normalize tokens
  k_norm<<<dim3(B_ * T_), dim3(256), 0, stream>>>(x, an_bf, bn_bf);
  // 3. similarity GEMM
  k_sim<<<dim3(TA_ / 16, TB_ / 16, B_), dim3(32), 0, stream>>>(an_bf, bn_bf, sim);
  // 4. greedy matching
  k_match<<<dim3(B_), dim3(256), 0, stream>>>(sim, pa, pb, kept, bmap);
  // 5. merged sequence
  k_build_merged<<<dim3(TM_, B_), dim3(256), 0, stream>>>(x, kept, merged);
  k_merge_pairs<<<dim3(R_, B_), dim3(256), 0, stream>>>(x, pa, pb, merged);
  // 6. Q/K/V projections (Q pre-scaled by 1/sqrt(hd)); V stored transposed
  {
    dim3 g(B_ * TM_ / (16 * GM_), D_ / (16 * GN_)), blk(32);
    k_gemm<<<g, blk, 0, stream>>>(merged, Wq_bf, nullptr, Qbf,  1, 0.125f);
    k_gemm<<<g, blk, 0, stream>>>(merged, Wk_bf, nullptr, Kbf,  1, 1.0f);
    k_gemm<<<g, blk, 0, stream>>>(merged, Wv_bf, nullptr, VTbf, 2, 1.0f);
  }
  // 7. attention
  k_attn<<<dim3(TM_ / 128, H_, B_), dim3(256), 0, stream>>>(Qbf, Kbf, VTbf, Obf);
  // 8. output projection
  k_gemm<<<dim3(B_ * TM_ / (16 * GM_), D_ / (16 * GN_)), dim3(32), 0, stream>>>(
      Obf, Wo_bf, Yf, nullptr, 0, 1.0f);
  // 9. unmerge
  k_unmerge<<<dim3(T_, B_), dim3(256), 0, stream>>>(Yf, bmap, out);
}
